// ImageEncoder_60095182406192
// MI455X (gfx1250) — compile-verified, matrix operand order checked by reading, not run on hardware
//
#include <hip/hip_runtime.h>
#include <math.h>

// ---------------- CDNA5 WMMA plumbing (gfx1250, wave32) ----------------
typedef __attribute__((ext_vector_type(16))) __bf16          v16bf;
typedef __attribute__((ext_vector_type(8)))  float           v8f;
typedef __attribute__((ext_vector_type(8)))  unsigned short  v8us;

__device__ __forceinline__ v16bf mkfrag(v8us lo, v8us hi) {
    union { struct { v8us lo, hi; } s; v16bf v; } u;
    u.s.lo = lo; u.s.hi = hi;
    return u.v;
}

__device__ __forceinline__ unsigned short f2bf(float f) {
    unsigned int u = __float_as_uint(f);
    unsigned int r = u + 0x7fffu + ((u >> 16) & 1u);   // round-to-nearest-even
    return (unsigned short)(r >> 16);
}

__device__ __forceinline__ unsigned int f2bf_pk(float lo, float hi) {
    return (unsigned int)f2bf(lo) | ((unsigned int)f2bf(hi) << 16);
}

__device__ __forceinline__ v8f wmma_bf16(v16bf a, v16bf b, v8f c) {
    return __builtin_amdgcn_wmma_f32_16x16x32_bf16(false, a, false, b, (short)0, c, false, false);
}

// 16-lane-group XOR-butterfly reductions. v_permlane16_b32 is pure VALU:
// no LDS round trip, no dscnt wait (vs ds_bpermute-based __shfl_xor).
#if __has_builtin(__builtin_amdgcn_permlane16)
__device__ __forceinline__ float pl16(float v, int s0, int s1) {
    return __int_as_float(__builtin_amdgcn_permlane16(
        __float_as_int(v), __float_as_int(v), s0, s1, false, false));
}
__device__ __forceinline__ float grp16_max(float v) {
    v = fmaxf(v, pl16(v, (int)0x67452301u, (int)0xEFCDAB89u));  // xor 1
    v = fmaxf(v, pl16(v, (int)0x54761032u, (int)0xDCFE98BAu));  // xor 2
    v = fmaxf(v, pl16(v, (int)0x32107654u, (int)0xBA98FEDCu));  // xor 4
    v = fmaxf(v, pl16(v, (int)0xFEDCBA98u, (int)0x76543210u));  // xor 8
    return v;
}
__device__ __forceinline__ float grp16_sum(float v) {
    v += pl16(v, (int)0x67452301u, (int)0xEFCDAB89u);
    v += pl16(v, (int)0x54761032u, (int)0xDCFE98BAu);
    v += pl16(v, (int)0x32107654u, (int)0xBA98FEDCu);
    v += pl16(v, (int)0xFEDCBA98u, (int)0x76543210u);
    return v;
}
#else
__device__ __forceinline__ float grp16_max(float v) {
    v = fmaxf(v, __shfl_xor(v, 1, 32));
    v = fmaxf(v, __shfl_xor(v, 2, 32));
    v = fmaxf(v, __shfl_xor(v, 4, 32));
    v = fmaxf(v, __shfl_xor(v, 8, 32));
    return v;
}
__device__ __forceinline__ float grp16_sum(float v) {
    v += __shfl_xor(v, 1, 32);
    v += __shfl_xor(v, 2, 32);
    v += __shfl_xor(v, 4, 32);
    v += __shfl_xor(v, 8, 32);
    return v;
}
#endif

#define NEG_INF (-__builtin_inff())

// ---------------- Model constants ----------------
#define BB   32
#define NTOK 197      // 196 patches + cls
#define NPAT 196
#define HH   768
#define NHEAD 12
#define HDIM 64
#define MLPD 3072
#define QKVD 2304

// =====================================================================
// Tiled GEMM:  out[M,N] = epilogue( A_bf16[M,K] @ W_f32[K,N] )
// TILE 32x128x64, 256 threads = 8 waves; each wave owns 2 N-tiles and
// runs 2 K-steps per stage -> 4 WMMAs per barrier pair. W is converted
// fp32 -> bf16 during LDS staging (each weight read exactly once per
// forward).
// ACT: 0 = none, 1 = exact GELU. res!=null: out = epi + res (fp32 h).
// =====================================================================
template <int ACT, bool OUTBF>
__global__ __launch_bounds__(256)
void gemm_wmma(const unsigned short* __restrict__ A, const float* __restrict__ W,
               const float* __restrict__ bias, const float* __restrict__ res,
               void* __restrict__ outp, int M, int K, int N)
{
    __shared__ unsigned short As[32][72];    // 144B row stride: 16B-aligned frags
    __shared__ unsigned short Bt[128][72];   // W tile, transposed (col-major)

    const int m0   = blockIdx.x * 32;
    const int n0   = blockIdx.y * 128;
    const int tid  = threadIdx.x;
    const int lane = tid & 31;
    const int wid  = tid >> 5;
    const int tm   = wid & 1;            // 2 M-tiles
    const int tnb  = (wid >> 1) * 2;     // base of this wave's 2 N-tiles
    const int lr   = lane & 15;
    const int kb   = (lane < 16) ? 0 : 8;      // A/B fragment K base
    const int hb   = (lane >> 4) << 3;         // C/D fragment M base

    v8f acc0 = {}, acc1 = {};

    const int arow = tid >> 3, ac8 = (tid & 7) * 8;      // A staging coords

    for (int k0 = 0; k0 < K; k0 += 64) {
        // stage A tile (32x64 bf16): one 16B load per thread
        {
            uint4 d = *(const uint4*)(A + (size_t)(m0 + arow) * K + k0 + ac8);
            *(uint4*)&As[arow][ac8] = d;
            if (k0 + 64 < K)
                __builtin_prefetch(A + (size_t)(m0 + arow) * K + k0 + 64 + ac8, 0, 3);
        }
        // stage W tile (64x128 fp32 -> bf16, transposed). Each chunk owns a
        // K-row pair x 8 N: packed b32 LDS stores (k is contiguous in Bt rows).
#pragma unroll
        for (int i = 0; i < 2; ++i) {
            int c  = tid + 256 * i;                // 512 chunks
            int nb = (c & 15) * 8;                 // 0..120, coalesced across tid
            int k2 = (c >> 4) * 2;                 // k-row pair 0,2,..,62
            const float* s0 = W + (size_t)(k0 + k2) * N + n0 + nb;
            const float* s1 = s0 + N;
            float4 a0 = ((const float4*)s0)[0];
            float4 a1 = ((const float4*)s0)[1];
            float4 b0 = ((const float4*)s1)[0];
            float4 b1 = ((const float4*)s1)[1];
            *(unsigned int*)&Bt[nb + 0][k2] = f2bf_pk(a0.x, b0.x);
            *(unsigned int*)&Bt[nb + 1][k2] = f2bf_pk(a0.y, b0.y);
            *(unsigned int*)&Bt[nb + 2][k2] = f2bf_pk(a0.z, b0.z);
            *(unsigned int*)&Bt[nb + 3][k2] = f2bf_pk(a0.w, b0.w);
            *(unsigned int*)&Bt[nb + 4][k2] = f2bf_pk(a1.x, b1.x);
            *(unsigned int*)&Bt[nb + 5][k2] = f2bf_pk(a1.y, b1.y);
            *(unsigned int*)&Bt[nb + 6][k2] = f2bf_pk(a1.z, b1.z);
            *(unsigned int*)&Bt[nb + 7][k2] = f2bf_pk(a1.w, b1.w);
            if (k0 + 64 < K)
                __builtin_prefetch(W + (size_t)(k0 + 64 + k2) * N + n0 + nb, 0, 3);
        }
        __syncthreads();

#pragma unroll
        for (int ks = 0; ks < 2; ++ks) {
            v16bf af = mkfrag(*(const v8us*)&As[tm * 16 + lr][ks * 32 + kb],
                              *(const v8us*)&As[tm * 16 + lr][ks * 32 + kb + 16]);
            v16bf b0 = mkfrag(*(const v8us*)&Bt[(tnb + 0) * 16 + lr][ks * 32 + kb],
                              *(const v8us*)&Bt[(tnb + 0) * 16 + lr][ks * 32 + kb + 16]);
            v16bf b1 = mkfrag(*(const v8us*)&Bt[(tnb + 1) * 16 + lr][ks * 32 + kb],
                              *(const v8us*)&Bt[(tnb + 1) * 16 + lr][ks * 32 + kb + 16]);
            acc0 = wmma_bf16(af, b0, acc0);
            acc1 = wmma_bf16(af, b1, acc1);
        }
        __syncthreads();
    }

    // epilogue (D layout: lane holds column n, rows hb..hb+7)
#pragma unroll
    for (int t = 0; t < 2; ++t) {
        const int n  = n0 + (tnb + t) * 16 + lr;
        const float bv = bias ? bias[n] : 0.0f;
#pragma unroll
        for (int r = 0; r < 8; ++r) {
            int m = m0 + tm * 16 + hb + r;
            float v = (t == 0 ? acc0[r] : acc1[r]) + bv;
            if (ACT == 1) v = 0.5f * v * (1.0f + erff(v * 0.70710678118f));
            if (res) v += res[(size_t)m * N + n];
            if (OUTBF) ((unsigned short*)outp)[(size_t)m * N + n] = f2bf(v);
            else       ((float*)outp)[(size_t)m * N + n]          = v;
        }
    }
}

// =====================================================================
// Flash-style attention: one wave per (query-16-tile, head, batch).
// S = Q K^T / 8 via WMMA (K rows contiguous -> direct b128 B-frag loads),
// online softmax (permlane16 butterflies), P via LDS transpose, and V
// staged transposed in LDS (coalesced b128 row loads instead of a
// per-lane u16 column gather from HBM) for the P@V WMMAs.
// =====================================================================
__global__ __launch_bounds__(32)
void attention_wmma(const unsigned short* __restrict__ qkv,
                    unsigned short* __restrict__ aout)
{
    __shared__ unsigned short Ps[16][40];   // P tile, A-layout staging
    __shared__ unsigned short Vt[64][40];   // V 32-key chunk, transposed [dim][key]

    const int mt   = blockIdx.x;   // 0..12 query tiles
    const int head = blockIdx.y;   // 0..11
    const int b    = blockIdx.z;   // 0..31
    const int lane = threadIdx.x & 31;
    const int lr   = lane & 15;
    const int hb   = (lane >> 4) << 3;
    const int kb   = (lane < 16) ? 0 : 8;
    const int qoff = head * HDIM;
    const int koff = HH + head * HDIM;
    const int voff = 2 * HH + head * HDIM;
    const int m0   = mt * 16;

    // Q A-fragments for both 32-wide K chunks of HD=64
    int qi = m0 + lr; if (qi > NTOK - 1) qi = NTOK - 1;
    const unsigned short* qp = qkv + (size_t)(b * NTOK + qi) * QKVD + qoff;
    v16bf qa0 = mkfrag(*(const v8us*)(qp + kb),      *(const v8us*)(qp + kb + 16));
    v16bf qa1 = mkfrag(*(const v8us*)(qp + 32 + kb), *(const v8us*)(qp + 32 + kb + 16));

    v8f o0 = {}, o1 = {}, o2 = {}, o3 = {};
    float mrow[8], lsum[8];
#pragma unroll
    for (int r = 0; r < 8; ++r) { mrow[r] = NEG_INF; lsum[r] = 0.0f; }

    for (int pt = 0; pt < 7; ++pt) {        // 7 pairs of 16-key tiles (13 real + 1 pad)
        // stage this pair's V chunk transposed: lane owns one key row,
        // 8 coalesced b128 loads + 64 b16 scatter stores (overlaps S WMMAs)
        {
            int key = pt * 32 + lane; if (key > NTOK - 1) key = NTOK - 1;
            const unsigned short* vp = qkv + (size_t)(b * NTOK + key) * QKVD + voff;
#pragma unroll
            for (int c = 0; c < 8; ++c) {
                v8us d8 = *(const v8us*)(vp + c * 8);
#pragma unroll
                for (int j = 0; j < 8; ++j) Vt[c * 8 + j][lane] = d8[j];
            }
        }

        v8f s0, s1;
        {   // key tile kt0 = 2*pt (always valid)
            int kt = pt * 2;
            int ki = kt * 16 + lr; int kig = ki > NTOK - 1 ? NTOK - 1 : ki;
            const unsigned short* kp = qkv + (size_t)(b * NTOK + kig) * QKVD + koff;
            v16bf k0f = mkfrag(*(const v8us*)(kp + kb),      *(const v8us*)(kp + kb + 16));
            v16bf k1f = mkfrag(*(const v8us*)(kp + 32 + kb), *(const v8us*)(kp + 32 + kb + 16));
            v8f z = {};
            s0 = wmma_bf16(qa0, k0f, z);
            s0 = wmma_bf16(qa1, k1f, s0);
            bool oob = ki > NTOK - 1;
#pragma unroll
            for (int r = 0; r < 8; ++r) s0[r] = oob ? NEG_INF : s0[r] * 0.125f;
        }
        {   // key tile kt1 = 2*pt+1 (tile 13 is padding)
            int kt = pt * 2 + 1;
            if (kt < 13) {
                int ki = kt * 16 + lr; int kig = ki > NTOK - 1 ? NTOK - 1 : ki;
                const unsigned short* kp = qkv + (size_t)(b * NTOK + kig) * QKVD + koff;
                v16bf k0f = mkfrag(*(const v8us*)(kp + kb),      *(const v8us*)(kp + kb + 16));
                v16bf k1f = mkfrag(*(const v8us*)(kp + 32 + kb), *(const v8us*)(kp + 32 + kb + 16));
                v8f z = {};
                s1 = wmma_bf16(qa0, k0f, z);
                s1 = wmma_bf16(qa1, k1f, s1);
                bool oob = ki > NTOK - 1;
#pragma unroll
                for (int r = 0; r < 8; ++r) s1[r] = oob ? NEG_INF : s1[r] * 0.125f;
            } else {
#pragma unroll
                for (int r = 0; r < 8; ++r) s1[r] = NEG_INF;
            }
        }

        // online softmax update (row r lives in a fixed 16-lane group)
        float p0a[8], p1a[8];
#pragma unroll
        for (int r = 0; r < 8; ++r) {
            float t  = grp16_max(fmaxf(s0[r], s1[r]));
            float mn = fmaxf(mrow[r], t);
            float sc = __expf(mrow[r] - mn);
            float p0 = __expf(s0[r] - mn);
            float p1 = __expf(s1[r] - mn);
            float rs = grp16_sum(p0 + p1);
            lsum[r]  = lsum[r] * sc + rs;
            mrow[r]  = mn;
            o0[r] *= sc; o1[r] *= sc; o2[r] *= sc; o3[r] *= sc;
            p0a[r] = p0; p1a[r] = p1;
        }

        // D-layout -> A-layout redistribution of P through LDS
#pragma unroll
        for (int r = 0; r < 8; ++r) {
            Ps[hb + r][lr]      = f2bf(p0a[r]);
            Ps[hb + r][16 + lr] = f2bf(p1a[r]);
        }
        asm volatile("s_wait_dscnt 0" ::: "memory");
        v16bf pa = mkfrag(*(const v8us*)&Ps[lr][kb], *(const v8us*)&Ps[lr][kb + 16]);

        // P(16x32) @ V(32x64): 4 N-tiles; B-frags from transposed LDS V
#pragma unroll
        for (int nt = 0; nt < 4; ++nt) {
            const int d = nt * 16 + lr;
            v16bf vb = mkfrag(*(const v8us*)&Vt[d][kb], *(const v8us*)&Vt[d][kb + 16]);
            if (nt == 0) o0 = wmma_bf16(pa, vb, o0);
            if (nt == 1) o1 = wmma_bf16(pa, vb, o1);
            if (nt == 2) o2 = wmma_bf16(pa, vb, o2);
            if (nt == 3) o3 = wmma_bf16(pa, vb, o3);
        }
    }

    // normalize + store bf16 into [B*N, H] at head*64 slice
#pragma unroll
    for (int r = 0; r < 8; ++r) {
        int tok = m0 + hb + r;
        if (tok < NTOK) {
            float inv = 1.0f / lsum[r];
            size_t base = (size_t)(b * NTOK + tok) * HH + head * HDIM;
            aout[base +  0 + lr] = f2bf(o0[r] * inv);
            aout[base + 16 + lr] = f2bf(o1[r] * inv);
            aout[base + 32 + lr] = f2bf(o2[r] * inv);
            aout[base + 48 + lr] = f2bf(o3[r] * inv);
        }
    }
}

// ---------------- LayerNorm helpers (one block = one 768-row) ----------------
__device__ __forceinline__ void block_reduce2(float& a, float& b_, float* sh) {
#pragma unroll
    for (int d = 16; d; d >>= 1) { a += __shfl_xor(a, d, 32); b_ += __shfl_xor(b_, d, 32); }
    int w = threadIdx.x >> 5, ln = threadIdx.x & 31;
    if (ln == 0) { sh[w * 2] = a; sh[w * 2 + 1] = b_; }
    __syncthreads();
    if (threadIdx.x == 0) {
        float s0 = 0, s1 = 0;
        for (int i = 0; i < (int)(blockDim.x >> 5); ++i) { s0 += sh[i * 2]; s1 += sh[i * 2 + 1]; }
        sh[0] = s0; sh[1] = s1;
    }
    __syncthreads();
    a = sh[0]; b_ = sh[1];
}

// patchify (b c (h p1) (w p2) -> (p1 p2 c)) + LN(pn1) -> bf16 A matrix
__global__ __launch_bounds__(256)
void patchify_ln(const float* __restrict__ x, const float* __restrict__ g,
                 const float* __restrict__ bt, unsigned short* __restrict__ out)
{
    __shared__ float sh[16];
    const int row = blockIdx.x;                 // b*196 + patch
    const int b = row / NPAT, patch = row % NPAT;
    const int gh = patch / 14, gw = patch % 14;
    float v[3];
#pragma unroll
    for (int i = 0; i < 3; ++i) {
        int e = threadIdx.x + i * 256;
        int c = e % 3, p2 = (e / 3) % 16, p1 = e / 48;
        v[i] = x[(((size_t)b * 3 + c) * 224 + gh * 16 + p1) * 224 + gw * 16 + p2];
    }
    float s = v[0] + v[1] + v[2];
    float q = v[0] * v[0] + v[1] * v[1] + v[2] * v[2];
    block_reduce2(s, q, sh);
    float mean = s * (1.0f / HH);
    float rstd = rsqrtf(q * (1.0f / HH) - mean * mean + 1e-5f);
#pragma unroll
    for (int i = 0; i < 3; ++i) {
        int e = threadIdx.x + i * 256;
        out[(size_t)row * HH + e] = f2bf((v[i] - mean) * rstd * g[e] + bt[e]);
    }
}

// generic LN(h)*g+b -> bf16
__global__ __launch_bounds__(256)
void ln_affine_bf16(const float* __restrict__ in, const float* __restrict__ g,
                    const float* __restrict__ bt, unsigned short* __restrict__ out)
{
    __shared__ float sh[16];
    const size_t row = blockIdx.x;
    float v[3];
#pragma unroll
    for (int i = 0; i < 3; ++i) v[i] = in[row * HH + threadIdx.x + i * 256];
    float s = v[0] + v[1] + v[2];
    float q = v[0] * v[0] + v[1] * v[1] + v[2] * v[2];
    block_reduce2(s, q, sh);
    float mean = s * (1.0f / HH);
    float rstd = rsqrtf(q * (1.0f / HH) - mean * mean + 1e-5f);
#pragma unroll
    for (int i = 0; i < 3; ++i) {
        int e = threadIdx.x + i * 256;
        out[row * HH + e] = f2bf((v[i] - mean) * rstd * g[e] + bt[e]);
    }
}

// h[b,0]=cls+pos[0]; h[b,t]=LN(p1[b,t-1])*g+b + pos[t]
__global__ __launch_bounds__(256)
void embed_kernel(const float* __restrict__ p1, const float* __restrict__ g,
                  const float* __restrict__ bt, const float* __restrict__ cls,
                  const float* __restrict__ pos, float* __restrict__ h)
{
    __shared__ float sh[16];
    const int row = blockIdx.x;                 // b*197 + tok
    const int b = row / NTOK, tok = row % NTOK;
    if (tok == 0) {
#pragma unroll
        for (int i = 0; i < 3; ++i) {
            int e = threadIdx.x + i * 256;
            h[(size_t)row * HH + e] = cls[e] + pos[e];
        }
        return;
    }
    const float* src = p1 + ((size_t)b * NPAT + tok - 1) * HH;
    float v[3];
#pragma unroll
    for (int i = 0; i < 3; ++i) v[i] = src[threadIdx.x + i * 256];
    float s = v[0] + v[1] + v[2];
    float q = v[0] * v[0] + v[1] * v[1] + v[2] * v[2];
    block_reduce2(s, q, sh);
    float mean = s * (1.0f / HH);
    float rstd = rsqrtf(q * (1.0f / HH) - mean * mean + 1e-5f);
#pragma unroll
    for (int i = 0; i < 3; ++i) {
        int e = threadIdx.x + i * 256;
        h[(size_t)row * HH + e] = (v[i] - mean) * rstd * g[e] + bt[e]
                                + pos[(size_t)tok * HH + e];
    }
}

__global__ void mean_pool(const float* __restrict__ h, float* __restrict__ out)
{
    const int b = blockIdx.x, e = threadIdx.x;      // 768 threads
    float s = 0.0f;
    for (int n = 0; n < NTOK; ++n) s += h[((size_t)b * NTOK + n) * HH + e];
    out[(size_t)b * HH + e] = s * (1.0f / NTOK);
}

// =====================================================================
extern "C" void kernel_launch(void* const* d_in, const int* in_sizes, int n_in,
                              void* d_out, int out_size, void* d_ws, size_t ws_size,
                              hipStream_t stream)
{
    const float* x      = (const float*)d_in[0];
    const float* pn1_g  = (const float*)d_in[1];
    const float* pn1_b  = (const float*)d_in[2];
    const float* pW     = (const float*)d_in[3];
    const float* pb     = (const float*)d_in[4];
    const float* pn2_g  = (const float*)d_in[5];
    const float* pn2_b  = (const float*)d_in[6];
    const float* cls    = (const float*)d_in[7];
    const float* pos    = (const float*)d_in[8];
    const float* ln1_g  = (const float*)d_in[9];
    const float* ln1_b  = (const float*)d_in[10];
    const float* qkv_W  = (const float*)d_in[11];
    const float* out_W  = (const float*)d_in[12];
    const float* ln2_g  = (const float*)d_in[13];
    const float* ln2_b  = (const float*)d_in[14];
    const float* fc1_W  = (const float*)d_in[15];
    const float* fc1_b  = (const float*)d_in[16];
    const float* fc2_W  = (const float*)d_in[17];
    const float* fc2_b  = (const float*)d_in[18];

    const size_t ROWS_P = (size_t)BB * NPAT;   // 6272
    const size_t ROWS_T = (size_t)BB * NTOK;   // 6304

    char* w = (char*)d_ws;
    auto carve = [&](size_t bytes) -> void* {
        void* p = (void*)w;
        w += (bytes + 255) & ~(size_t)255;
        return p;
    };
    unsigned short* p_ln  = (unsigned short*)carve(ROWS_P * HH   * 2);
    float*          p1    = (float*)         carve(ROWS_P * HH   * 4);
    float*          h     = (float*)         carve(ROWS_T * HH   * 4);
    unsigned short* y     = (unsigned short*)carve(ROWS_T * HH   * 2);
    unsigned short* qkvb  = (unsigned short*)carve(ROWS_T * QKVD * 2);
    unsigned short* attnb = (unsigned short*)carve(ROWS_T * HH   * 2);
    unsigned short* mlp1  = (unsigned short*)carve(ROWS_T * MLPD * 2);
    (void)ws_size; (void)in_sizes; (void)n_in; (void)out_size;

    // patch embedding
    patchify_ln<<<dim3(ROWS_P), 256, 0, stream>>>(x, pn1_g, pn1_b, p_ln);
    gemm_wmma<0, false><<<dim3(196, HH / 128), 256, 0, stream>>>(
        p_ln, pW, pb, nullptr, p1, ROWS_P, HH, HH);
    embed_kernel<<<dim3(ROWS_T), 256, 0, stream>>>(p1, pn2_g, pn2_b, cls, pos, h);

    // transformer layers
    for (int l = 0; l < 12; ++l) {
        ln_affine_bf16<<<dim3(ROWS_T), 256, 0, stream>>>(
            h, ln1_g + l * HH, ln1_b + l * HH, y);
        gemm_wmma<0, true><<<dim3(197, QKVD / 128), 256, 0, stream>>>(
            y, qkv_W + (size_t)l * HH * QKVD, nullptr, nullptr, qkvb, ROWS_T, HH, QKVD);
        attention_wmma<<<dim3(13, NHEAD, BB), 32, 0, stream>>>(qkvb, attnb);
        gemm_wmma<0, false><<<dim3(197, HH / 128), 256, 0, stream>>>(
            attnb, out_W + (size_t)l * HH * HH, nullptr, h, h, ROWS_T, HH, HH);
        ln_affine_bf16<<<dim3(ROWS_T), 256, 0, stream>>>(
            h, ln2_g + l * HH, ln2_b + l * HH, y);
        gemm_wmma<1, true><<<dim3(197, MLPD / 128), 256, 0, stream>>>(
            y, fc1_W + (size_t)l * HH * MLPD, fc1_b + l * MLPD, nullptr, mlp1, ROWS_T, HH, MLPD);
        gemm_wmma<1, false><<<dim3(197, HH / 128), 256, 0, stream>>>(
            mlp1, fc2_W + (size_t)l * MLPD * HH, fc2_b + l * HH, h, h, ROWS_T, MLPD, HH);
    }

    mean_pool<<<dim3(BB), 768, 0, stream>>>(h, (float*)d_out);
}
